// MLAttention_72198400246398
// MI455X (gfx1250) — matrix operand order repaired and checked by reading, not compiled, not run on hardware
//
#include <hip/hip_runtime.h>
#include <hip/hip_bf16.h>

// ---------------------------------------------------------------------------
// MLA forward for MI455X (gfx1250, wave32, WMMA bf16 16x16x32, f32 accum)
// GEMMs: TDM (tensor_load_to_lds) double-buffered LDS staging + WMMA.
// ---------------------------------------------------------------------------

#define T_SEQ   3072
#define DIM     2048
#define N_HEADS 16
#define Q_LORA  1536
#define KV_LORA 512
#define NOPE    128
#define ROPE    64
#define V_DIM   128
#define DQK     192   // NOPE + ROPE

typedef __attribute__((ext_vector_type(16))) __bf16 v16bf;
typedef __attribute__((ext_vector_type(8)))  __bf16 v8bf;
typedef __attribute__((ext_vector_type(8)))  float  v8f;
typedef __attribute__((ext_vector_type(4)))  unsigned int v4u;
typedef __attribute__((ext_vector_type(8)))  int v8i;
typedef __attribute__((ext_vector_type(4)))  int v4i;

static __device__ __forceinline__ unsigned short f32_to_bf16_bits(float f) {
    unsigned int u = __float_as_uint(f);
    unsigned int r = u + 0x7FFFu + ((u >> 16) & 1u);   // round-to-nearest-even
    return (unsigned short)(r >> 16);
}

static __device__ __forceinline__ v8f wmma_bf16(v16bf a, v16bf b, v8f c) {
    return __builtin_amdgcn_wmma_f32_16x16x32_bf16(
        /*neg_a=*/false, a, /*neg_b=*/false, b,
        /*c_mod=*/(short)0, c, /*reuse_a=*/false, /*reuse_b=*/false);
}

// Load A fragment (16x32 bf16 slice, row-major, row stride `ldk` elems).
// A elem e -> K = e + 16*half: two contiguous 8-elem chunks.
static __device__ __forceinline__ v16bf load_a_frag(const unsigned short* row,
                                                    int k0, int half) {
    v8bf lo = *(const v8bf*)(row + k0 + 16 * half);
    v8bf hi = *(const v8bf*)(row + k0 + 16 * half + 8);
    v16bf a;
#pragma unroll
    for (int i = 0; i < 8; ++i) { a[i] = lo[i]; a[8 + i] = hi[i]; }
    return a;
}

// ---------------------------------------------------------------------------
// TDM: issue a 2-D tile load (bf16 elements) global -> LDS.
// D# per CDNA5 ISA ch.8: group0 = {count, lds_addr, global_addr, type=2},
// group1 = {data_size=1(2B), tensor dims == tile dims, tile 2-D, dim0 stride}.
// ---------------------------------------------------------------------------
#if __has_builtin(__builtin_amdgcn_tensor_load_to_lds)
#define USE_TDM 1
static __device__ __forceinline__ void tdm_load_2d_bf16(
    const unsigned short* gsrc, unsigned lds_byte_addr,
    int tile_d0 /*k elems*/, int tile_d1 /*rows*/, int row_stride_elems) {
    unsigned long long ga = (unsigned long long)gsrc;
    v4u g0;
    g0[0] = 1u;                                        // count=1, user desc
    g0[1] = lds_byte_addr;                             // lds_addr
    g0[2] = (unsigned)(ga & 0xFFFFFFFFu);              // global_addr[31:0]
    g0[3] = (unsigned)((ga >> 32) & 0x01FFFFFFu) | (2u << 30);  // [56:32] | type=2
    unsigned long long d0 = (unsigned)tile_d0;         // tensor_dim0 == tile_dim0
    unsigned long long d1 = (unsigned)tile_d1;         // tensor_dim1 == tile_dim1
    unsigned long long s0 = (unsigned long long)(unsigned)row_stride_elems;
    unsigned long long qa = (1ull << 16)               // data_size = 1 (2 bytes)
                          | ((d0 & 0xFFFFull) << 48);  // tensor_dim0[15:0]
    unsigned long long qb = ((d0 >> 16) & 0xFFFFull)   // tensor_dim0[31:16]
                          | ((d1 & 0xFFFFFFFFull) << 16)          // tensor_dim1
                          | (((unsigned long long)tile_d0 & 0xFFFFull) << 48);
    unsigned long long qc = ((unsigned long long)tile_d1 & 0xFFFFull)  // tile_dim1
                          /* tile_dim2 = 0 (unused) */
                          | ((s0 & 0xFFFFFFFFull) << 32);   // dim0_stride[31:0]
    unsigned long long qd = ((s0 >> 32) & 0xFFFFull);       // dim0_stride[47:32]
    v8i g1;
    g1[0] = (int)(qa & 0xFFFFFFFFu); g1[1] = (int)(qa >> 32);
    g1[2] = (int)(qb & 0xFFFFFFFFu); g1[3] = (int)(qb >> 32);
    g1[4] = (int)(qc & 0xFFFFFFFFu); g1[5] = (int)(qc >> 32);
    g1[6] = (int)(qd & 0xFFFFFFFFu); g1[7] = (int)(qd >> 32);
    v4i z4 = {0, 0, 0, 0};
#if defined(__clang_major__) && (__clang_major__ >= 23)
    v8i z8 = {0, 0, 0, 0, 0, 0, 0, 0};
    __builtin_amdgcn_tensor_load_to_lds(g0, g1, z4, z4, z8, 0);
#else
    __builtin_amdgcn_tensor_load_to_lds(g0, g1, z4, z4, 0);
#endif
}
#else
#define USE_TDM 0
#endif

// ---------------------------------------------------------------------------
// Elementwise conversions
// ---------------------------------------------------------------------------
__global__ void conv_bf16(const float* __restrict__ in, unsigned short* __restrict__ out,
                          long long n) {
    long long i = (long long)blockIdx.x * blockDim.x + threadIdx.x;
    if (i < n) out[i] = f32_to_bf16_bits(in[i]);
}

// in: [K,N] fp32 row-major -> out: [N,K] bf16 row-major (weight transpose)
__global__ void conv_bf16_t(const float* __restrict__ in, unsigned short* __restrict__ out,
                            int K, int N) {
    int n = blockIdx.x * 16 + threadIdx.x;
    int k = blockIdx.y * 16 + threadIdx.y;
    if (k < K && n < N) out[(size_t)n * K + k] = f32_to_bf16_bits(in[(size_t)k * N + n]);
}

// ---------------------------------------------------------------------------
// WMMA GEMM: C[M,N] (f32) = A[M,K] (bf16 row-major) * Bt[N,K]^T (bf16)
// Block = 256 threads = 8 waves (4 M-waves x 2 N-waves); block tile 64x64.
// TDM stages 64x64 K-panels of A and Bt into double-buffered LDS; wave 0
// drives the DMA; compute waves read fragments via ds_load.
// Requires M%64==0, N%64==0, K%64==0 (true for all GEMMs here).
// ---------------------------------------------------------------------------
__global__ __launch_bounds__(256) void gemm_bf16_wmma(
    const unsigned short* __restrict__ A,
    const unsigned short* __restrict__ Bt,
    float* __restrict__ C, int M, int N, int K) {
    const int lane = threadIdx.x & 31;
    const int wave = threadIdx.x >> 5;
    const int half = lane >> 4;
    const int l16  = lane & 15;
    const int mw   = wave >> 1;             // 0..3
    const int nw   = wave & 1;              // 0..1
    const int mblk = blockIdx.y * 64;
    const int nblk = blockIdx.x * 64;
    const int m0 = mblk + mw * 16;
    const int n0 = nblk + nw * 32;

    v8f acc0 = {}; v8f acc1 = {};

#if USE_TDM
    __shared__ __align__(16) unsigned short lds_a[2][64 * 64];
    __shared__ __align__(16) unsigned short lds_b[2][64 * 64];
    const int nstages = K >> 6;             // K / 64
    if (wave == 0) {
        tdm_load_2d_bf16(A  + (size_t)mblk * K, (unsigned)(unsigned long long)&lds_a[0][0],
                         64, 64, K);
        tdm_load_2d_bf16(Bt + (size_t)nblk * K, (unsigned)(unsigned long long)&lds_b[0][0],
                         64, 64, K);
    }
    for (int s = 0; s < nstages; ++s) {
        const int cur = s & 1;
        if (wave == 0) {
            if (s + 1 < nstages) {
                int k1 = (s + 1) << 6;
                tdm_load_2d_bf16(A  + (size_t)mblk * K + k1,
                                 (unsigned)(unsigned long long)&lds_a[cur ^ 1][0], 64, 64, K);
                tdm_load_2d_bf16(Bt + (size_t)nblk * K + k1,
                                 (unsigned)(unsigned long long)&lds_b[cur ^ 1][0], 64, 64, K);
                __builtin_amdgcn_s_wait_tensorcnt(2);   // stage s panels resident
            } else {
                __builtin_amdgcn_s_wait_tensorcnt(0);
            }
        }
        __syncthreads();
        const unsigned short* arow  = &lds_a[cur][(mw * 16 + l16) * 64];
        const unsigned short* b0row = &lds_b[cur][(nw * 32 + l16) * 64];
        const unsigned short* b1row = b0row + 16 * 64;
#pragma unroll
        for (int kc = 0; kc < 64; kc += 32) {
            v16bf a  = load_a_frag(arow, kc, half);
            v16bf b0 = *(const v16bf*)(b0row + kc + 16 * half);
            v16bf b1 = *(const v16bf*)(b1row + kc + 16 * half);
            acc0 = wmma_bf16(a, b0, acc0);
            acc1 = wmma_bf16(a, b1, acc1);
        }
        __syncthreads();                    // buffer being DMA'd next is free
    }
#else
    const unsigned short* arow  = A  + (size_t)(m0 + l16) * K;
    const unsigned short* b0row = Bt + (size_t)(n0 + l16) * K;
    const unsigned short* b1row = Bt + (size_t)(n0 + 16 + l16) * K;
    for (int k0 = 0; k0 < K; k0 += 32) {
        v16bf a  = load_a_frag(arow, k0, half);
        v16bf b0 = *(const v16bf*)(b0row + k0 + 16 * half);
        v16bf b1 = *(const v16bf*)(b1row + k0 + 16 * half);
        acc0 = wmma_bf16(a, b0, acc0);
        acc1 = wmma_bf16(a, b1, acc1);
    }
#endif
#pragma unroll
    for (int r = 0; r < 8; ++r) {
        int row = m0 + r + 8 * half;
        C[(size_t)row * N + n0 + l16]      = acc0[r];
        C[(size_t)row * N + n0 + 16 + l16] = acc1[r];
    }
}

// ---------------------------------------------------------------------------
// Row RMSNorm (fp32 in -> bf16 out), one block (256 thr) per row
// ---------------------------------------------------------------------------
__global__ __launch_bounds__(256) void rmsnorm_bf16(
    const float* __restrict__ in, const float* __restrict__ w,
    unsigned short* __restrict__ out, int N) {
    __shared__ float red[256];
    int row = blockIdx.x;
    const float* x = in + (size_t)row * N;
    float ss = 0.f;
    for (int i = threadIdx.x; i < N; i += 256) { float v = x[i]; ss += v * v; }
    red[threadIdx.x] = ss; __syncthreads();
    for (int s = 128; s > 0; s >>= 1) {
        if ((int)threadIdx.x < s) red[threadIdx.x] += red[threadIdx.x + s];
        __syncthreads();
    }
    float inv = rsqrtf(red[0] / (float)N + 1e-6f);
    for (int i = threadIdx.x; i < N; i += 256)
        out[(size_t)row * N + i] = f32_to_bf16_bits(x[i] * inv * w[i]);
}

// ---------------------------------------------------------------------------
// KV prep: rmsnorm(kvd[:,0:512]) -> ckv_bf ; RoPE(kvd[:,512:576]) -> kropeb
// ---------------------------------------------------------------------------
__global__ __launch_bounds__(256) void kv_prep(
    const float* __restrict__ kvd, const float* __restrict__ kvw,
    const float* __restrict__ freqs,
    unsigned short* __restrict__ ckv, unsigned short* __restrict__ kropeb) {
    __shared__ float red[256];
    int t = blockIdx.x;
    const float* x = kvd + (size_t)t * (KV_LORA + ROPE);
    float ss = 0.f;
    for (int i = threadIdx.x; i < KV_LORA; i += 256) { float v = x[i]; ss += v * v; }
    red[threadIdx.x] = ss; __syncthreads();
    for (int s = 128; s > 0; s >>= 1) {
        if ((int)threadIdx.x < s) red[threadIdx.x] += red[threadIdx.x + s];
        __syncthreads();
    }
    float inv = rsqrtf(red[0] / (float)KV_LORA + 1e-6f);
    for (int i = threadIdx.x; i < KV_LORA; i += 256)
        ckv[(size_t)t * KV_LORA + i] = f32_to_bf16_bits(x[i] * inv * kvw[i]);
    if (threadIdx.x < ROPE / 2) {
        int i = threadIdx.x;
        float fr = freqs[(size_t)t * (ROPE / 2) + i];
        float c = __cosf(fr), s = __sinf(fr);
        float x1 = x[KV_LORA + 2 * i], x2 = x[KV_LORA + 2 * i + 1];
        kropeb[(size_t)t * ROPE + 2 * i]     = f32_to_bf16_bits(x1 * c - x2 * s);
        kropeb[(size_t)t * ROPE + 2 * i + 1] = f32_to_bf16_bits(x1 * s + x2 * c);
    }
}

// ---------------------------------------------------------------------------
// Assemble per-head bf16 Q [h][t][192], K [h][t][192], V^T [h][d][t]
// ---------------------------------------------------------------------------
__global__ __launch_bounds__(256) void assemble_qkv(
    const float* __restrict__ q_nope, const float* __restrict__ q_rope_raw,
    const float* __restrict__ kv_up, const unsigned short* __restrict__ kropeb,
    const float* __restrict__ freqs,
    unsigned short* __restrict__ Qb, unsigned short* __restrict__ Kb,
    unsigned short* __restrict__ Vtb) {
    int t = blockIdx.x;
    int tid = threadIdx.x;
    for (int h = 0; h < N_HEADS; ++h) {
        size_t qk_base = ((size_t)h * T_SEQ + t) * DQK;
        if (tid < NOPE) {
            Qb[qk_base + tid] = f32_to_bf16_bits(q_nope[(size_t)t * (N_HEADS * NOPE) + h * NOPE + tid]);
            Kb[qk_base + tid] = f32_to_bf16_bits(kv_up[(size_t)t * (N_HEADS * 256) + h * 256 + tid]);
            Vtb[((size_t)h * V_DIM + tid) * T_SEQ + t] =
                f32_to_bf16_bits(kv_up[(size_t)t * (N_HEADS * 256) + h * 256 + NOPE + tid]);
        } else if (tid < NOPE + ROPE / 2) {
            int i = tid - NOPE;
            float fr = freqs[(size_t)t * (ROPE / 2) + i];
            float c = __cosf(fr), s = __sinf(fr);
            float x1 = q_rope_raw[(size_t)t * (N_HEADS * ROPE) + h * ROPE + 2 * i];
            float x2 = q_rope_raw[(size_t)t * (N_HEADS * ROPE) + h * ROPE + 2 * i + 1];
            Qb[qk_base + NOPE + 2 * i]     = f32_to_bf16_bits(x1 * c - x2 * s);
            Qb[qk_base + NOPE + 2 * i + 1] = f32_to_bf16_bits(x1 * s + x2 * c);
        } else if (tid < NOPE + ROPE / 2 + ROPE) {
            int j = tid - NOPE - ROPE / 2;
            Kb[qk_base + NOPE + j] = kropeb[(size_t)t * ROPE + j];
        }
    }
}

// ---------------------------------------------------------------------------
// Causal flash attention: one wave per (16-row q-tile, head).
// ---------------------------------------------------------------------------
__global__ __launch_bounds__(32) void mla_flash(
    const unsigned short* __restrict__ Qb,   // [H][T][192]
    const unsigned short* __restrict__ Kb,   // [H][T][192]
    const unsigned short* __restrict__ Vtb,  // [H][128][T]
    unsigned short* __restrict__ Ob) {       // [T][H*128] bf16
    __shared__ __align__(32) unsigned short plds[16 * 32];
    const int lane = threadIdx.x & 31;
    const int half = lane >> 4;
    const int l16  = lane & 15;
    const int q0 = blockIdx.x * 16;
    const int h  = blockIdx.y;

    const unsigned short* Qh = Qb  + (size_t)h * T_SEQ * DQK;
    const unsigned short* Kh = Kb  + (size_t)h * T_SEQ * DQK;
    const unsigned short* Vh = Vtb + (size_t)h * V_DIM * T_SEQ;

    v16bf qf[6];
    {
        const unsigned short* qrow = Qh + (size_t)(q0 + l16) * DQK;
#pragma unroll
        for (int kc = 0; kc < 6; ++kc) qf[kc] = load_a_frag(qrow, kc * 32, half);
    }

    float m_i[8], l_i[8];
#pragma unroll
    for (int r = 0; r < 8; ++r) { m_i[r] = -1e30f; l_i[r] = 0.f; }
    v8f o[8];
#pragma unroll
    for (int ct = 0; ct < 8; ++ct) o[ct] = (v8f){};

    const float scale = rsqrtf((float)DQK);
    const int nblk = (q0 + 16 + 31) >> 5;

    for (int b = 0; b < nblk; ++b) {
        const int k0 = b * 32;
        v8f s[2];
#pragma unroll
        for (int c = 0; c < 2; ++c) {
            v8f acc = {};
            const unsigned short* krow = Kh + (size_t)(k0 + c * 16 + l16) * DQK;
#pragma unroll
            for (int kc = 0; kc < 6; ++kc) {
                v16bf bf = *(const v16bf*)(krow + kc * 32 + 16 * half);
                acc = wmma_bf16(qf[kc], bf, acc);
            }
            s[c] = acc;
        }
        float mnew[8], alpha[8];
#pragma unroll
        for (int r = 0; r < 8; ++r) {
            int row = q0 + r + 8 * half;
            float v0 = s[0][r] * scale;
            float v1 = s[1][r] * scale;
            if (k0 + l16 > row)      v0 = -1e30f;
            if (k0 + 16 + l16 > row) v1 = -1e30f;
            s[0][r] = v0; s[1][r] = v1;
            float mx = fmaxf(v0, v1);
#pragma unroll
            for (int off = 1; off < 16; off <<= 1) mx = fmaxf(mx, __shfl_xor(mx, off, 32));
            mnew[r] = fmaxf(m_i[r], mx);
        }
#pragma unroll
        for (int r = 0; r < 8; ++r) {
            float p0 = __expf(s[0][r] - mnew[r]);
            float p1 = __expf(s[1][r] - mnew[r]);
            s[0][r] = p0; s[1][r] = p1;
            float rs = p0 + p1;
#pragma unroll
            for (int off = 1; off < 16; off <<= 1) rs += __shfl_xor(rs, off, 32);
            alpha[r] = __expf(m_i[r] - mnew[r]);
            l_i[r] = l_i[r] * alpha[r] + rs;
            m_i[r] = mnew[r];
        }
#pragma unroll
        for (int r = 0; r < 8; ++r) {
            int row = r + 8 * half;
            plds[row * 32 + l16]      = f32_to_bf16_bits(s[0][r]);
            plds[row * 32 + 16 + l16] = f32_to_bf16_bits(s[1][r]);
        }
        asm volatile("s_wait_dscnt 0" ::: "memory");
        v16bf pa;
        {
            const unsigned short* prow = plds + l16 * 32;
            v8bf lo = *(const v8bf*)(prow + 16 * half);
            v8bf hi = *(const v8bf*)(prow + 16 * half + 8);
#pragma unroll
            for (int i = 0; i < 8; ++i) { pa[i] = lo[i]; pa[8 + i] = hi[i]; }
        }
        asm volatile("" ::: "memory");
#pragma unroll
        for (int ct = 0; ct < 8; ++ct) {
            v8f oc = o[ct];
#pragma unroll
            for (int r = 0; r < 8; ++r) oc[r] *= alpha[r];
            v16bf vb = *(const v16bf*)(Vh + (size_t)(ct * 16 + l16) * T_SEQ + k0 + 16 * half);
            o[ct] = wmma_bf16(pa, vb, oc);
        }
    }
#pragma unroll
    for (int ct = 0; ct < 8; ++ct) {
#pragma unroll
        for (int r = 0; r < 8; ++r) {
            int row = q0 + r + 8 * half;
            int col = ct * 16 + l16;
            Ob[(size_t)row * (N_HEADS * V_DIM) + h * V_DIM + col] =
                f32_to_bf16_bits(o[ct][r] / l_i[r]);
        }
    }
}

// ---------------------------------------------------------------------------
// Host launch
// ---------------------------------------------------------------------------
extern "C" void kernel_launch(void* const* d_in, const int* in_sizes, int n_in,
                              void* d_out, int out_size, void* d_ws, size_t ws_size,
                              hipStream_t stream) {
    (void)in_sizes; (void)n_in; (void)out_size; (void)ws_size;
    const float* x         = (const float*)d_in[0];
    const float* freqs     = (const float*)d_in[1];
    /* d_in[2] = mask: causal handled analytically */
    const float* wq_down   = (const float*)d_in[3];
    const float* q_norm_w  = (const float*)d_in[4];
    const float* wq_nope   = (const float*)d_in[5];
    const float* wq_rope   = (const float*)d_in[6];
    const float* wkv_down  = (const float*)d_in[7];
    const float* kv_norm_w = (const float*)d_in[8];
    const float* wkv_up    = (const float*)d_in[9];
    const float* wo        = (const float*)d_in[10];
    float* out = (float*)d_out;

    char* p = (char*)d_ws;
    auto carve = [&](size_t bytes) -> void* {
        void* r = (void*)p;
        p += (bytes + 255) & ~(size_t)255;
        return r;
    };
    unsigned short* xb      = (unsigned short*)carve((size_t)T_SEQ * DIM * 2);
    unsigned short* wqd_t   = (unsigned short*)carve((size_t)Q_LORA * DIM * 2);
    unsigned short* wqn_t   = (unsigned short*)carve((size_t)(N_HEADS * NOPE) * Q_LORA * 2);
    unsigned short* wqr_t   = (unsigned short*)carve((size_t)(N_HEADS * ROPE) * Q_LORA * 2);
    unsigned short* wkvd_t  = (unsigned short*)carve((size_t)(KV_LORA + ROPE) * DIM * 2);
    unsigned short* wkvu_t  = (unsigned short*)carve((size_t)(N_HEADS * 256) * KV_LORA * 2);
    unsigned short* wo_t    = (unsigned short*)carve((size_t)DIM * (N_HEADS * V_DIM) * 2);
    unsigned short* qlat_b  = (unsigned short*)carve((size_t)T_SEQ * Q_LORA * 2);
    unsigned short* ckv_b   = (unsigned short*)carve((size_t)T_SEQ * KV_LORA * 2);
    unsigned short* kropeb  = (unsigned short*)carve((size_t)T_SEQ * ROPE * 2);
    unsigned short* Qb      = (unsigned short*)carve((size_t)N_HEADS * T_SEQ * DQK * 2);
    unsigned short* Kb      = (unsigned short*)carve((size_t)N_HEADS * T_SEQ * DQK * 2);
    unsigned short* Vtb     = (unsigned short*)carve((size_t)N_HEADS * V_DIM * T_SEQ * 2);
    unsigned short* attn_b  = (unsigned short*)carve((size_t)T_SEQ * (N_HEADS * V_DIM) * 2);
    float* qlat_raw = (float*)carve((size_t)T_SEQ * Q_LORA * 4);
    float* kvd      = (float*)carve((size_t)T_SEQ * (KV_LORA + ROPE) * 4);
    float* q_nope   = (float*)carve((size_t)T_SEQ * (N_HEADS * NOPE) * 4);
    float* q_rope_r = (float*)carve((size_t)T_SEQ * (N_HEADS * ROPE) * 4);
    float* kv_up    = (float*)carve((size_t)T_SEQ * (N_HEADS * 256) * 4);

    dim3 tb16(16, 16);
    {
        long long n = (long long)T_SEQ * DIM;
        conv_bf16<<<(unsigned)((n + 255) / 256), 256, 0, stream>>>(x, xb, n);
    }
    conv_bf16_t<<<dim3(Q_LORA / 16, DIM / 16), tb16, 0, stream>>>(wq_down, wqd_t, DIM, Q_LORA);
    conv_bf16_t<<<dim3((N_HEADS * NOPE) / 16, Q_LORA / 16), tb16, 0, stream>>>(wq_nope, wqn_t, Q_LORA, N_HEADS * NOPE);
    conv_bf16_t<<<dim3((N_HEADS * ROPE) / 16, Q_LORA / 16), tb16, 0, stream>>>(wq_rope, wqr_t, Q_LORA, N_HEADS * ROPE);
    conv_bf16_t<<<dim3((KV_LORA + ROPE) / 16, DIM / 16), tb16, 0, stream>>>(wkv_down, wkvd_t, DIM, KV_LORA + ROPE);
    conv_bf16_t<<<dim3((N_HEADS * 256) / 16, KV_LORA / 16), tb16, 0, stream>>>(wkv_up, wkvu_t, KV_LORA, N_HEADS * 256);
    conv_bf16_t<<<dim3(DIM / 16, (N_HEADS * V_DIM) / 16), tb16, 0, stream>>>(wo, wo_t, N_HEADS * V_DIM, DIM);

    gemm_bf16_wmma<<<dim3(Q_LORA / 64, T_SEQ / 64), 256, 0, stream>>>(xb, wqd_t, qlat_raw, T_SEQ, Q_LORA, DIM);
    gemm_bf16_wmma<<<dim3((KV_LORA + ROPE) / 64, T_SEQ / 64), 256, 0, stream>>>(xb, wkvd_t, kvd, T_SEQ, KV_LORA + ROPE, DIM);

    rmsnorm_bf16<<<T_SEQ, 256, 0, stream>>>(qlat_raw, q_norm_w, qlat_b, Q_LORA);
    kv_prep<<<T_SEQ, 256, 0, stream>>>(kvd, kv_norm_w, freqs, ckv_b, kropeb);

    gemm_bf16_wmma<<<dim3((N_HEADS * NOPE) / 64, T_SEQ / 64), 256, 0, stream>>>(qlat_b, wqn_t, q_nope, T_SEQ, N_HEADS * NOPE, Q_LORA);
    gemm_bf16_wmma<<<dim3((N_HEADS * ROPE) / 64, T_SEQ / 64), 256, 0, stream>>>(qlat_b, wqr_t, q_rope_r, T_SEQ, N_HEADS * ROPE, Q_LORA);
    gemm_bf16_wmma<<<dim3((N_HEADS * 256) / 64, T_SEQ / 64), 256, 0, stream>>>(ckv_b, wkvu_t, kv_up, T_SEQ, N_HEADS * 256, KV_LORA);

    assemble_qkv<<<T_SEQ, 256, 0, stream>>>(q_nope, q_rope_r, kv_up, kropeb, freqs, Qb, Kb, Vtb);
    mla_flash<<<dim3(T_SEQ / 16, N_HEADS), 32, 0, stream>>>(Qb, Kb, Vtb, attn_b);

    gemm_bf16_wmma<<<dim3(DIM / 64, T_SEQ / 64), 256, 0, stream>>>(attn_b, wo_t, out, T_SEQ, DIM, N_HEADS * V_DIM);
}
